// TransformerXL_55319178773051
// MI455X (gfx1250) — compile-verified, matrix operand order checked by reading, not run on hardware
//
#include <hip/hip_runtime.h>
#include <math.h>

typedef __bf16 bf16;
typedef __attribute__((ext_vector_type(16))) __bf16 v16bf;
typedef __attribute__((ext_vector_type(8)))  __bf16 v8bf;
typedef __attribute__((ext_vector_type(8)))  float   v8f;

#define Tt   2048
#define Bb   2
#define Dm   512
#define Hh   8
#define Dh   64
#define Li   4
#define Din  2048
#define INCH 2048
#define NCLS 64
#define MTB  (Tt*Bb)   // 4096 rows (t-major, b inner)

// ---------------------------------------------------------------------------
// WMMA helpers (bf16 A/B, f32 accumulate), CDNA5 16x16x32 shape.
// A fragment: lane = M row (lane&15), K packed: halves hold K kb..kb+7 and
// kb+16..kb+23 with kb = (lane>>4)*8.  B fragment identical with lane = N col,
// reading a [N,K] row-major matrix (i.e. computes A * W^T).
// ---------------------------------------------------------------------------
__device__ __forceinline__ v16bf frag_combine(v8bf lo, v8bf hi) {
  v16bf f;
#pragma unroll
  for (int i = 0; i < 8; ++i) { f[i] = lo[i]; f[i + 8] = hi[i]; }
  return f;
}

__device__ __forceinline__ v16bf load_frag(const bf16* __restrict__ base, int ld,
                                           int row0, int k0, int lane) {
  const bf16* p = base + (size_t)(row0 + (lane & 15)) * ld + k0 + ((lane >> 4) << 3);
  v8bf lo = *(const v8bf*)p;
  v8bf hi = *(const v8bf*)(p + 16);
  return frag_combine(lo, hi);
}

__device__ __forceinline__ v8f wmma_bf16(v16bf a, v16bf b, v8f c) {
  return __builtin_amdgcn_wmma_f32_16x16x32_bf16(false, a, false, b, (short)0, c,
                                                 false, false);
}

// ---------------------------------------------------------------------------
// Generic GEMM: C[M,N] = A[M,K](bf16) * W[N,K](bf16)^T (+bias)(+resid)(relu)
// block = 256 thr = 8 waves (4x2), block tile 128(M) x 64(N), wave tile 32x32.
// M%128==0, N%64==0, K%32==0 for all call sites.
// ---------------------------------------------------------------------------
__global__ __launch_bounds__(256) void gemm_bf16_kernel(
    const bf16* __restrict__ A, const bf16* __restrict__ W,
    float* __restrict__ C, const float* __restrict__ bias,
    const float* __restrict__ resid, bf16* __restrict__ Cbf,
    int M, int N, int K, int relu) {
  const int lane = threadIdx.x & 31;
  const int wave = threadIdx.x >> 5;
  const int m0 = blockIdx.x * 128 + (wave >> 1) * 32;
  const int n0 = blockIdx.y * 64 + (wave & 1) * 32;

  v8f acc[2][2];
#pragma unroll
  for (int i = 0; i < 2; ++i)
#pragma unroll
    for (int j = 0; j < 2; ++j) acc[i][j] = (v8f){0, 0, 0, 0, 0, 0, 0, 0};

  for (int k0 = 0; k0 < K; k0 += 32) {
    if (k0 + 32 < K) {
      __builtin_prefetch(A + (size_t)(m0 + (lane & 15)) * K + k0 + 32, 0, 1);
      __builtin_prefetch(W + (size_t)(n0 + (lane & 15)) * K + k0 + 32, 0, 1);
    }
    v16bf a0 = load_frag(A, K, m0, k0, lane);
    v16bf a1 = load_frag(A, K, m0 + 16, k0, lane);
    v16bf b0 = load_frag(W, K, n0, k0, lane);
    v16bf b1 = load_frag(W, K, n0 + 16, k0, lane);
    acc[0][0] = wmma_bf16(a0, b0, acc[0][0]);
    acc[0][1] = wmma_bf16(a0, b1, acc[0][1]);
    acc[1][0] = wmma_bf16(a1, b0, acc[1][0]);
    acc[1][1] = wmma_bf16(a1, b1, acc[1][1]);
  }

  const int half = lane >> 4, tn = lane & 15;
#pragma unroll
  for (int mt = 0; mt < 2; ++mt)
#pragma unroll
    for (int nt = 0; nt < 2; ++nt) {
      int nn = n0 + nt * 16 + tn;
      float bv = bias ? bias[nn] : 0.0f;
#pragma unroll
      for (int r = 0; r < 8; ++r) {
        int mm = m0 + mt * 16 + r + 8 * half;
        size_t idx = (size_t)mm * N + nn;
        float v = acc[mt][nt][r] + bv;
        if (resid) v += resid[idx];
        if (relu) v = fmaxf(v, 0.0f);
        C[idx] = v;
        if (Cbf) Cbf[idx] = (bf16)v;
      }
    }
}

// ---------------------------------------------------------------------------
// Flash attention with Transformer-XL relative-position band.
// block = 128 thr = 4 waves; wave owns 16 q rows; streams 32-wide k tiles.
// ---------------------------------------------------------------------------
__global__ __launch_bounds__(128) void attn_kernel(
    const bf16* __restrict__ qw_buf, const bf16* __restrict__ qr_buf,
    const bf16* __restrict__ k_buf, const bf16* __restrict__ vT_buf,
    const bf16* __restrict__ rho_buf, bf16* __restrict__ av_buf) {
  __shared__ float pband_s[4][16 * 48];
  __shared__ __attribute__((aligned(16))) bf16 ptile_s[4][16 * 32];

  const int lane = threadIdx.x & 31;
  const int wave = threadIdx.x >> 5;
  const int half = lane >> 4, tj = lane & 15;
  const int qbase = blockIdx.x * 64 + wave * 16;
  const int bn = blockIdx.y;             // b*8 + n
  const int b = bn >> 3, n = bn & 7;

  const bf16* qw = qw_buf + (size_t)bn * Tt * Dh;
  const bf16* qr = qr_buf + (size_t)bn * Tt * Dh;
  const bf16* kp = k_buf + (size_t)bn * Tt * Dh;
  const bf16* vT = vT_buf + (size_t)bn * Dh * Tt;
  const bf16* rho = rho_buf + (size_t)n * Tt * Dh;
  float* pband = pband_s[wave];
  bf16* ptile = ptile_s[wave];

  const v16bf aqw0 = load_frag(qw, Dh, qbase, 0, lane);
  const v16bf aqw1 = load_frag(qw, Dh, qbase, 32, lane);
  const v16bf aqr0 = load_frag(qr, Dh, qbase, 0, lane);
  const v16bf aqr1 = load_frag(qr, Dh, qbase, 32, lane);

  v8f oacc[4];
#pragma unroll
  for (int d = 0; d < 4; ++d) oacc[d] = (v8f){0, 0, 0, 0, 0, 0, 0, 0};
  float rmax[8], rsum[8];
#pragma unroll
  for (int r = 0; r < 8; ++r) { rmax[r] = -3.0e38f; rsum[r] = 0.0f; }
  const float scale = 0.125f;  // 1/sqrt(64)

  for (int j0 = 0; j0 < Tt; j0 += 32) {
    // ---- content term AC = (q + r_w_bias) . K^T  (two 16-col tiles) ----
    v8f s0 = (v8f){0, 0, 0, 0, 0, 0, 0, 0};
    v8f s1 = (v8f){0, 0, 0, 0, 0, 0, 0, 0};
    s0 = wmma_bf16(aqw0, load_frag(kp, Dh, j0, 0, lane), s0);
    s0 = wmma_bf16(aqw1, load_frag(kp, Dh, j0, 32, lane), s0);
    s1 = wmma_bf16(aqw0, load_frag(kp, Dh, j0 + 16, 0, lane), s1);
    s1 = wmma_bf16(aqw1, load_frag(kp, Dh, j0 + 16, 32, lane), s1);

    // ---- position term: band product P[m,c] = q̂_m . rho[pbase+c] ----
    const int pbase = qbase - j0 - 31;   // covers i-j over the 32-wide k tile
#pragma unroll
    for (int ct = 0; ct < 3; ++ct) {
      int p = pbase + ct * 16 + tj;
      p = (p < 0) ? 0 : ((p > Tt - 1) ? Tt - 1 : p);
      const bf16* rp = rho + (size_t)p * Dh + (half << 3);
      v16bf br0 = frag_combine(*(const v8bf*)rp, *(const v8bf*)(rp + 16));
      v16bf br1 = frag_combine(*(const v8bf*)(rp + 32), *(const v8bf*)(rp + 48));
      v8f pt = (v8f){0, 0, 0, 0, 0, 0, 0, 0};
      pt = wmma_bf16(aqr0, br0, pt);
      pt = wmma_bf16(aqr1, br1, pt);
#pragma unroll
      for (int r = 0; r < 8; ++r)
        pband[(r + 8 * half) * 48 + ct * 16 + tj] = pt[r];
    }
    asm volatile("s_wait_dscnt 0x0" ::: "memory");

    // ---- S = (AC + BD(rel-shift)) * scale; online softmax ----
    float tmax[8];
#pragma unroll
    for (int r = 0; r < 8; ++r) {
      int m = r + 8 * half;
      float v0 = (s0[r] + pband[m * 48 + (m - tj + 31)]) * scale;
      float v1 = (s1[r] + pband[m * 48 + (m - tj + 15)]) * scale;
      s0[r] = v0; s1[r] = v1;
      tmax[r] = fmaxf(v0, v1);
    }
#pragma unroll
    for (int off = 1; off < 16; off <<= 1)
#pragma unroll
      for (int r = 0; r < 8; ++r)
        tmax[r] = fmaxf(tmax[r], __shfl_xor(tmax[r], off, 32));

    float corr[8], psum[8];
#pragma unroll
    for (int r = 0; r < 8; ++r) {
      float nm = fmaxf(rmax[r], tmax[r]);
      corr[r] = __expf(rmax[r] - nm);
      rmax[r] = nm;
      float e0 = __expf(s0[r] - nm);
      float e1 = __expf(s1[r] - nm);
      int m = r + 8 * half;
      ptile[m * 32 + tj] = (bf16)e0;
      ptile[m * 32 + 16 + tj] = (bf16)e1;
      psum[r] = e0 + e1;
    }
#pragma unroll
    for (int off = 1; off < 16; off <<= 1)
#pragma unroll
      for (int r = 0; r < 8; ++r) psum[r] += __shfl_xor(psum[r], off, 32);
#pragma unroll
    for (int r = 0; r < 8; ++r) rsum[r] = rsum[r] * corr[r] + psum[r];
#pragma unroll
    for (int d = 0; d < 4; ++d)
#pragma unroll
      for (int r = 0; r < 8; ++r) oacc[d][r] *= corr[r];

    asm volatile("s_wait_dscnt 0x0" ::: "memory");
    // reload exp(S) as an A fragment (16x32, K = key index)
    const bf16* pp = ptile + (size_t)tj * 32 + (half << 3);
    v16bf pa = frag_combine(*(const v8bf*)pp, *(const v8bf*)(pp + 16));
#pragma unroll
    for (int dt = 0; dt < 4; ++dt) {
      const bf16* vp = vT + (size_t)(dt * 16 + tj) * Tt + j0 + (half << 3);
      v16bf bv = frag_combine(*(const v8bf*)vp, *(const v8bf*)(vp + 16));
      oacc[dt] = wmma_bf16(pa, bv, oacc[dt]);
    }
  }

  // ---- normalize and emit attn_vec rows (bf16, [t*B+b, n*64+d]) ----
#pragma unroll
  for (int dt = 0; dt < 4; ++dt)
#pragma unroll
    for (int r = 0; r < 8; ++r) {
      int m = r + 8 * half;
      int t = qbase + m;
      float o = oacc[dt][r] / rsum[r];
      av_buf[((size_t)(t * Bb + b)) * (Hh * Dh) + n * Dh + dt * 16 + tj] = (bf16)o;
    }
}

// ---------------------------------------------------------------------------
// LayerNorm over D=512: one block (256 thr) per row; writes f32 + bf16 mirror.
// ---------------------------------------------------------------------------
__global__ __launch_bounds__(256) void ln_kernel(
    const float* __restrict__ in, const float* __restrict__ sc,
    const float* __restrict__ bi, float* __restrict__ outf,
    bf16* __restrict__ outbf, int D) {
  __shared__ float wsum[8], wsq[8], mu_s, rs_s;
  const int lane = threadIdx.x & 31, wave = threadIdx.x >> 5;
  const float* x = in + (size_t)blockIdx.x * D;
  float ls = 0.0f, lq = 0.0f;
  for (int i = threadIdx.x; i < D; i += 256) {
    float v = x[i];
    ls += v; lq += v * v;
  }
#pragma unroll
  for (int off = 1; off < 32; off <<= 1) {
    ls += __shfl_xor(ls, off, 32);
    lq += __shfl_xor(lq, off, 32);
  }
  if (lane == 0) { wsum[wave] = ls; wsq[wave] = lq; }
  __syncthreads();
  if (threadIdx.x == 0) {
    float s = 0, q = 0;
    for (int w = 0; w < 8; ++w) { s += wsum[w]; q += wsq[w]; }
    float mu = s / D;
    float var = q / D - mu * mu;
    mu_s = mu;
    rs_s = rsqrtf(var + 1e-5f);
  }
  __syncthreads();
  float mu = mu_s, rs = rs_s;
  for (int i = threadIdx.x; i < D; i += 256) {
    float v = (x[i] - mu) * rs * sc[i] + bi[i];
    size_t o = (size_t)blockIdx.x * D + i;
    outf[o] = v;
    outbf[o] = (bf16)v;
  }
}

// ---------------------------------------------------------------------------
// Elementwise / packing kernels
// ---------------------------------------------------------------------------
__global__ void cvt_bf16_kernel(const float* __restrict__ in, bf16* __restrict__ out,
                                size_t n) {
  for (size_t i = (size_t)blockIdx.x * blockDim.x + threadIdx.x; i < n;
       i += (size_t)gridDim.x * blockDim.x)
    out[i] = (bf16)in[i];
}

__global__ void xpose_kernel(const float* __restrict__ x, bf16* __restrict__ xbf) {
  const size_t total = (size_t)MTB * INCH;
  for (size_t i = (size_t)blockIdx.x * blockDim.x + threadIdx.x; i < total;
       i += (size_t)gridDim.x * blockDim.x) {
    int c = (int)(i % INCH);
    size_t m = i / INCH;
    int t = (int)(m >> 1), b = (int)(m & 1);
    xbf[i] = (bf16)x[((size_t)b * INCH + c) * Tt + t];
  }
}

__global__ void posemb_kernel(bf16* __restrict__ pos) {
  const size_t total = (size_t)Tt * Dm;
  for (size_t i = (size_t)blockIdx.x * blockDim.x + threadIdx.x; i < total;
       i += (size_t)gridDim.x * blockDim.x) {
    int d = (int)(i % Dm);
    int t = (int)(i / Dm);
    int j = (d < 256) ? d : d - 256;
    float invf = __expf(-(2.0f * j / 512.0f) * 9.210340372f);  // ln(10000)
    float a = (float)(Tt - 1 - t) * invf;
    pos[i] = (bf16)((d < 256) ? __sinf(a) : __cosf(a));
  }
}

// qkv C[M,1536] (bias already applied) -> qw/qr/k [b,n][T,64], vT [b,n][64,T]
__global__ void qkv_rearrange_kernel(const float* __restrict__ C,
                                     const float* __restrict__ rwb,
                                     const float* __restrict__ rrb,
                                     bf16* __restrict__ qw, bf16* __restrict__ qr,
                                     bf16* __restrict__ kk, bf16* __restrict__ vT) {
  const size_t total = (size_t)MTB * (3 * Hh * Dh);
  for (size_t i = (size_t)blockIdx.x * blockDim.x + threadIdx.x; i < total;
       i += (size_t)gridDim.x * blockDim.x) {
    int e = (int)(i % (3 * Hh * Dh));
    size_t m = i / (3 * Hh * Dh);
    int t = (int)(m >> 1), b = (int)(m & 1);
    int part = e / (Hh * Dh);
    int n = (e % (Hh * Dh)) / Dh;
    int d = e % Dh;
    float v = C[i];
    size_t o = ((size_t)(b * Hh + n) * Tt + t) * Dh + d;
    if (part == 0) {
      qw[o] = (bf16)(v + rwb[n * Dh + d]);
      qr[o] = (bf16)(v + rrb[n * Dh + d]);
    } else if (part == 1) {
      kk[o] = (bf16)v;
    } else {
      vT[((size_t)(b * Hh + n) * Dh + d) * Tt + t] = (bf16)v;
    }
  }
}

// r_head_k C[T,512] -> rho[n][p][d] with reversed position index
__global__ void rk_rearrange_kernel(const float* __restrict__ C,
                                    bf16* __restrict__ rho) {
  const size_t total = (size_t)Hh * Tt * Dh;
  for (size_t i = (size_t)blockIdx.x * blockDim.x + threadIdx.x; i < total;
       i += (size_t)gridDim.x * blockDim.x) {
    int n = (int)(i / ((size_t)Tt * Dh));
    int rem = (int)(i % ((size_t)Tt * Dh));
    int p = rem / Dh, d = rem % Dh;
    rho[i] = (bf16)C[(size_t)(Tt - 1 - p) * Dm + n * Dh + d];
  }
}

// cls C[M,64] (bias applied) -> out[b, k, t]
__global__ void cls_out_kernel(const float* __restrict__ C, float* __restrict__ out) {
  const size_t total = (size_t)MTB * NCLS;
  for (size_t i = (size_t)blockIdx.x * blockDim.x + threadIdx.x; i < total;
       i += (size_t)gridDim.x * blockDim.x) {
    int kc = (int)(i % NCLS);
    size_t m = i / NCLS;
    int t = (int)(m >> 1), b = (int)(m & 1);
    out[((size_t)b * NCLS + kc) * Tt + t] = C[i];
  }
}

// ---------------------------------------------------------------------------
// Host orchestration
// ---------------------------------------------------------------------------
extern "C" void kernel_launch(void* const* d_in, const int* in_sizes, int n_in,
                              void* d_out, int out_size, void* d_ws, size_t ws_size,
                              hipStream_t stream) {
  (void)in_sizes; (void)n_in; (void)out_size; (void)ws_size;
  const float* x     = (const float*)d_in[0];
  const float* emb_w = (const float*)d_in[1];
  const float* emb_b = (const float*)d_in[2];
  const float* rwb   = (const float*)d_in[3];
  const float* rrb   = (const float*)d_in[4];
  const float* qkv_w = (const float*)d_in[5];
  const float* qkv_b = (const float*)d_in[6];
  const float* rp_w  = (const float*)d_in[7];
  const float* o_w   = (const float*)d_in[8];
  const float* ln1s  = (const float*)d_in[9];
  const float* ln1b  = (const float*)d_in[10];
  const float* f1_w  = (const float*)d_in[11];
  const float* f1_b  = (const float*)d_in[12];
  const float* f2_w  = (const float*)d_in[13];
  const float* f2_b  = (const float*)d_in[14];
  const float* ln2s  = (const float*)d_in[15];
  const float* ln2b  = (const float*)d_in[16];
  const float* cls_w = (const float*)d_in[17];
  const float* cls_b = (const float*)d_in[18];
  float* out = (float*)d_out;

  size_t off = 0;
  auto wsalloc = [&](size_t bytes) -> void* {
    void* p = (char*)d_ws + off;
    off += (bytes + 255) & ~(size_t)255;
    return p;
  };
  bf16*  hbf  = (bf16*)wsalloc((size_t)MTB * Dm * 2);
  float* hF   = (float*)wsalloc((size_t)MTB * Dm * 4);
  float* tmpF = (float*)wsalloc((size_t)MTB * Dm * 4);
  bf16*  xbf  = (bf16*)wsalloc((size_t)MTB * INCH * 2);
  float* Cbuf = (float*)wsalloc((size_t)MTB * Din * 4);
  bf16*  fbf  = (bf16*)wsalloc((size_t)MTB * Din * 2);
  bf16*  qwB  = (bf16*)wsalloc((size_t)Bb * Hh * Tt * Dh * 2);
  bf16*  qrB  = (bf16*)wsalloc((size_t)Bb * Hh * Tt * Dh * 2);
  bf16*  kB   = (bf16*)wsalloc((size_t)Bb * Hh * Tt * Dh * 2);
  bf16*  vTB  = (bf16*)wsalloc((size_t)Bb * Hh * Dh * Tt * 2);
  bf16*  rhoB = (bf16*)wsalloc((size_t)Hh * Tt * Dh * 2);
  bf16*  avB  = (bf16*)wsalloc((size_t)MTB * Dm * 2);
  bf16*  posB = (bf16*)wsalloc((size_t)Tt * Dm * 2);
  bf16*  wEmb = (bf16*)wsalloc((size_t)Dm * INCH * 2);
  bf16*  wQkv = (bf16*)wsalloc((size_t)Li * 3 * Hh * Dh * Dm * 2);
  bf16*  wRp  = (bf16*)wsalloc((size_t)Li * Dm * Dm * 2);
  bf16*  wO   = (bf16*)wsalloc((size_t)Li * Dm * Dm * 2);
  bf16*  wF1  = (bf16*)wsalloc((size_t)Li * Din * Dm * 2);
  bf16*  wF2  = (bf16*)wsalloc((size_t)Li * Dm * Din * 2);
  bf16*  wCls = (bf16*)wsalloc((size_t)NCLS * Dm * 2);

  const dim3 eb(256);
  const dim3 eg(2048);
  auto cvt = [&](const float* src, bf16* dst, size_t n) {
    cvt_bf16_kernel<<<eg, eb, 0, stream>>>(src, dst, n);
  };
  cvt(emb_w, wEmb, (size_t)Dm * INCH);
  cvt(qkv_w, wQkv, (size_t)Li * 3 * Hh * Dh * Dm);
  cvt(rp_w, wRp, (size_t)Li * Dm * Dm);
  cvt(o_w, wO, (size_t)Li * Dm * Dm);
  cvt(f1_w, wF1, (size_t)Li * Din * Dm);
  cvt(f2_w, wF2, (size_t)Li * Dm * Din);
  cvt(cls_w, wCls, (size_t)NCLS * Dm);

  xpose_kernel<<<eg, eb, 0, stream>>>(x, xbf);
  posemb_kernel<<<eg, eb, 0, stream>>>(posB);

  auto gemm = [&](const bf16* A, const bf16* W, float* C, const float* bias,
                  const float* resid, bf16* Cbf, int M, int N, int K, int relu) {
    dim3 g(M / 128, N / 64);
    gemm_bf16_kernel<<<g, 256, 0, stream>>>(A, W, C, bias, resid, Cbf, M, N, K, relu);
  };

  // embedding: h = xbf * emb_w^T + emb_b     [4096, 512]
  gemm(xbf, wEmb, hF, emb_b, nullptr, hbf, MTB, Dm, INCH, 0);

  for (int i = 0; i < Li; ++i) {
    // QKV projection [4096, 1536]
    gemm(hbf, wQkv + (size_t)i * 3 * Hh * Dh * Dm, Cbuf, qkv_b + (size_t)i * 3 * Hh * Dh,
         nullptr, nullptr, MTB, 3 * Hh * Dh, Dm, 0);
    qkv_rearrange_kernel<<<eg, eb, 0, stream>>>(Cbuf, rwb, rrb, qwB, qrB, kB, vTB);

    // r_head_k = pos_emb * r_proj^T  [2048, 512] -> rho (reversed index)
    gemm(posB, wRp + (size_t)i * Dm * Dm, Cbuf, nullptr, nullptr, nullptr, Tt, Dm, Dm, 0);
    rk_rearrange_kernel<<<eg, eb, 0, stream>>>(Cbuf, rhoB);

    // flash attention -> attn_vec (bf16 rows)
    attn_kernel<<<dim3(Tt / 64, Bb * Hh), 128, 0, stream>>>(qwB, qrB, kB, vTB, rhoB, avB);

    // output projection + residual, then LN1
    gemm(avB, wO + (size_t)i * Dm * Dm, tmpF, nullptr, hF, nullptr, MTB, Dm, Dm, 0);
    ln_kernel<<<MTB, 256, 0, stream>>>(tmpF, ln1s + (size_t)i * Dm, ln1b + (size_t)i * Dm,
                                       hF, hbf, Dm);

    // FFN
    gemm(hbf, wF1 + (size_t)i * Din * Dm, Cbuf, f1_b + (size_t)i * Din, nullptr, fbf,
         MTB, Din, Dm, 1);
    gemm(fbf, wF2 + (size_t)i * Dm * Din, tmpF, f2_b + (size_t)i * Dm, hF, nullptr,
         MTB, Dm, Din, 0);
    ln_kernel<<<MTB, 256, 0, stream>>>(tmpF, ln2s + (size_t)i * Dm, ln2b + (size_t)i * Dm,
                                       hF, hbf, Dm);
  }

  // classifier -> [B, 64, T]
  gemm(hbf, wCls, Cbuf, cls_b, nullptr, nullptr, MTB, NCLS, Dm, 0);
  cls_out_kernel<<<eg, eb, 0, stream>>>(Cbuf, out);
}
